// H2GCN_88802743812566
// MI455X (gfx1250) — matrix operand order repaired and checked by reading, not compiled, not run on hardware
//
#include <hip/hip_runtime.h>
#include <hip/hip_bf16.h>

#define NNODES 50000
#define NPAD   (NNODES + 64)
#define NEDGES 1600000
#define IN_CH  512
#define HID    256
#define OUT_CH 40
#define BN_EPS 1e-5f

typedef __bf16 v16bf __attribute__((ext_vector_type(16)));
typedef float  v8f   __attribute__((ext_vector_type(8)));
typedef unsigned int u32x4 __attribute__((ext_vector_type(4)));
typedef int          i32x4 __attribute__((ext_vector_type(4)));
typedef int          i32x8 __attribute__((ext_vector_type(8)));

#if defined(__has_builtin)
#if __has_builtin(__builtin_amdgcn_tensor_load_to_lds) && __has_builtin(__builtin_amdgcn_s_wait_tensorcnt)
#define USE_TDM 1
#endif
#endif

__device__ __forceinline__ __bf16 f2bf(float f) {
    unsigned u = __float_as_uint(f);
    unsigned short s = (unsigned short)((u + 0x7FFFu + ((u >> 16) & 1u)) >> 16);
    return __builtin_bit_cast(__bf16, s);
}

// ---------------- small utility kernels ----------------

__global__ void k_zero(float* __restrict__ p, long long n) {
    long long i = (long long)blockIdx.x * blockDim.x + threadIdx.x;
    if (i < n) p[i] = 0.0f;
}

__global__ void k_f2bf_arr(const float* __restrict__ src, unsigned short* __restrict__ dst, int n) {
    int i = blockIdx.x * blockDim.x + threadIdx.x;
    if (i < n) dst[i] = __builtin_bit_cast(unsigned short, f2bf(src[i]));
}

// convert [K x ncol] fp32 into [K x ldpad] bf16 (pad columns pre-zeroed by caller)
__global__ void k_f2bf_pad(const float* __restrict__ src, unsigned short* __restrict__ dst,
                           int ncol, int ldpad, int total /* K*ncol */) {
    int i = blockIdx.x * blockDim.x + threadIdx.x;
    if (i < total) {
        int k = i / ncol, c = i % ncol;
        dst[k * ldpad + c] = __builtin_bit_cast(unsigned short, f2bf(src[i]));
    }
}

__global__ void k_degree(const int* __restrict__ row, float* __restrict__ deg, int E) {
    int e = blockIdx.x * blockDim.x + threadIdx.x;
    if (e < E) atomicAdd(&deg[row[e]], 1.0f);
}

__global__ void k_dis(const float* __restrict__ deg, float* __restrict__ dis, int n) {
    int i = blockIdx.x * blockDim.x + threadIdx.x;
    if (i < n) dis[i] = rsqrtf(fmaxf(deg[i], 1.0f));
}

__global__ void k_norm(const int* __restrict__ row, const int* __restrict__ col,
                       const float* __restrict__ dis, float* __restrict__ nrm, int E) {
    int e = blockIdx.x * blockDim.x + threadIdx.x;
    if (e < E) nrm[e] = dis[row[e]] * dis[col[e]];
}

// zero combined[:, 256:768] (agg slices), combined stride 768
__global__ void k_zero_agg(float* __restrict__ comb, long long total /* N*512 */) {
    long long i = (long long)blockIdx.x * blockDim.x + threadIdx.x;
    if (i < total) {
        long long r = i / 512;
        int c = (int)(i % 512);
        comb[r * 768 + 256 + c] = 0.0f;
    }
}

// ---------------- SpMM: agg[col] += norm * src[row], feature dim 256, stride 768 ----------------

__global__ __launch_bounds__(256) void k_spmm(const float* __restrict__ srcfeat,
                                              float* __restrict__ dstfeat,
                                              const int* __restrict__ row,
                                              const int* __restrict__ col,
                                              const float* __restrict__ nrm, int E) {
    int e = blockIdx.x * 8 + (threadIdx.x >> 5);
    if (e >= E) return;
    int lane = threadIdx.x & 31;
    int s = row[e];
    int d = col[e];
    float w = nrm[e];
    const float* hs = srcfeat + (size_t)s * 768;
    float* hd = dstfeat + (size_t)d * 768;
#pragma unroll
    for (int i = 0; i < 8; ++i) {
        int f = lane + 32 * i;
        atomicAdd(&hd[f], w * hs[f]);
    }
}

// ---------------- BatchNorm ----------------

__global__ __launch_bounds__(256) void k_bn_reduce(const float* __restrict__ y,
                                                   float* __restrict__ bsum,
                                                   float* __restrict__ bss, int Nn) {
    int c = threadIdx.x;
    int r0 = blockIdx.x * 128;
    int r1 = r0 + 128; if (r1 > Nn) r1 = Nn;
    float s = 0.0f, ss = 0.0f;
    for (int r = r0; r < r1; ++r) {
        float v = y[(size_t)r * HID + c];
        s += v;
        ss += v * v;
    }
    atomicAdd(&bsum[c], s);
    atomicAdd(&bss[c], ss);
}

__global__ void k_bn_final(const float* __restrict__ bsum, const float* __restrict__ bss,
                           const float* __restrict__ g, const float* __restrict__ beta,
                           float* __restrict__ scale, float* __restrict__ shift, int Nn) {
    int c = threadIdx.x;
    float inv_n = 1.0f / (float)Nn;
    float mean = bsum[c] * inv_n;
    float var  = bss[c] * inv_n - mean * mean;
    float sc = g[c] * rsqrtf(var + BN_EPS);
    scale[c] = sc;
    shift[c] = beta[c] - mean * sc;
}

__global__ void k_bn_apply(const float* __restrict__ y, const float* __restrict__ scale,
                           const float* __restrict__ shift, float* __restrict__ comb,
                           long long total /* N*256 */) {
    long long i = (long long)blockIdx.x * blockDim.x + threadIdx.x;
    if (i < total) {
        int c = (int)(i % HID);
        long long r = i / HID;
        float v = y[i] * scale[c] + shift[c];
        comb[r * 768 + c] = fmaxf(v, 0.0f);
    }
}

// fp32 combined -> bf16 combined mirror (GEMM A operand, halves HBM traffic vs fp32 re-reads)
__global__ void k_comb2bf(const float* __restrict__ comb, unsigned short* __restrict__ combbf,
                          long long total /* N*768 */) {
    long long i = (long long)blockIdx.x * blockDim.x + threadIdx.x;
    if (i < total) combbf[i] = __builtin_bit_cast(unsigned short, f2bf(comb[i]));
}

// ---------------- bf16 WMMA GEMM: C = [relu](A_bf16 @ B_bf16 + bias) ----------------
// block = 256 threads = 8 wave32s; tile BM=64 BN=64 BK=32
// A tile staged via Tensor Data Mover (TDM) with LDS padding (row stride 40 bf16) when available.
// B panels are column-padded so every 64-wide block is fully covered: no guards before WMMA.

#define BM 64
#define BN 64
#define BK 32
#define ASTR 40   // bf16 elems/row in LDS: 32 data + 8 pad (TDM pad: every 16 dwords add 4 dwords)
#define BSTR 40

__global__ __launch_bounds__(256) void k_gemm_bf16(const unsigned short* __restrict__ Aus, int lda,
                                                   const unsigned short* __restrict__ Bus, int ldb,
                                                   const float* __restrict__ bias,
                                                   float* __restrict__ C, int ldc,
                                                   int M, int Nt, int K, int do_relu) {
    __shared__ __bf16 sA[BM * ASTR];       // [row][k], stride 40
    __shared__ __bf16 sB[BN * BSTR];       // transposed: [n][k]

    const int tid  = threadIdx.x;
    const int m0   = blockIdx.x * BM;
    const int n0   = blockIdx.y * BN;
    const int lane = tid & 31;
    const int wave = tid >> 5;
    const int wm   = wave & 3;
    const int wn   = wave >> 2;
    const int half = lane >> 4;
    const int l16  = lane & 15;
    const __bf16* Ah = (const __bf16*)Aus;
    const __bf16* Bh = (const __bf16*)Bus;

    v8f acc0 = {};
    v8f acc1 = {};

#ifdef USE_TDM
    // constant parts of the tensor descriptor (D#)
    const unsigned lds_base = (unsigned)(unsigned long long)(void*)&sA[0];
    u32x4 g0;
    g0[0] = 1u;                                   // count=1, user mode, no gather
    g0[1] = lds_base;                             // lds_addr
    i32x8 g1;
    g1[0] = (int)((1u << 16)                      // data_size = 1 -> 2 bytes
                | (1u << 20)                      // pad_enable
                | (3u << 22)                      // pad_interval: 16 dwords (one 32-elem bf16 row)
                | (3u << 25));                    // pad_amount: 4 dwords (8 bf16) -> row stride 40
    g1[1] = (int)(((unsigned)K & 0xFFFFu) << 16); // abar=0 | tensor_dim0[15:0]
    g1[2] = (int)((((unsigned)K >> 16) & 0xFFFFu) | (((unsigned)M & 0xFFFFu) << 16)); // dim0 hi | dim1 lo
    g1[3] = (int)((((unsigned)M >> 16) & 0xFFFFu) | ((unsigned)BK << 16));            // dim1 hi | tile_dim0
    g1[4] = (int)((unsigned)BM);                  // tile_dim1 = 64, tile_dim2 = 0
    g1[5] = (int)(unsigned)lda;                   // tensor_dim0_stride lo
    g1[6] = 0;                                    // stride hi | dim1_stride lo
    g1[7] = 0;
    i32x4 gz = {};
#if __has_include(<hip/amd_detail/amd_gfx1250_TDM.h>)
    i32x8 gz8 = {};
#endif
#endif

    for (int k0 = 0; k0 < K; k0 += BK) {
#ifdef USE_TDM
        if (wave == 0) {
            unsigned long long ga = (unsigned long long)(const void*)Ah
                                  + 2ull * ((unsigned long long)(unsigned)m0 * (unsigned)lda + (unsigned)k0);
            u32x4 g0i = g0;
            g0i[2] = (unsigned)ga;                                        // global_addr[31:0]
            g0i[3] = (unsigned)((ga >> 32) & 0x1FFFFFFull) | (2u << 30);  // global_addr[56:32] | type=2
#if __has_include(<hip/amd_detail/amd_gfx1250_TDM.h>)
            __builtin_amdgcn_tensor_load_to_lds(g0i, g1, gz, gz, gz8, 0);
#else
            __builtin_amdgcn_tensor_load_to_lds(g0i, g1, gz, gz, 0);
#endif
        }
#else
        // manual staging: 64 rows x 32 bf16, one 16B chunk per thread (A buffers are row-padded)
        {
            int r  = tid >> 2;              // 0..63
            int ch = tid & 3;               // 4 chunks of 8 bf16 per row
            const uint4* src = (const uint4*)&Ah[(size_t)(m0 + r) * lda + k0 + ch * 8];
            *(uint4*)&sA[r * ASTR + ch * 8] = *src;
        }
#endif
        // stage B tile (32xBN bf16): unconditional vector loads (B panels column-padded),
        // transpose into sB[n][k]
        {
            int kk = tid >> 3;              // 0..31
            int ch = tid & 7;               // 8 chunks of 8 bf16 per tile row
            uint4 pkt = *(const uint4*)&Bh[(size_t)(k0 + kk) * ldb + n0 + ch * 8];
            const __bf16* el = (const __bf16*)&pkt;
#pragma unroll
            for (int j = 0; j < 8; ++j) sB[(ch * 8 + j) * BSTR + kk] = el[j];
        }
#ifdef USE_TDM
        if (wave == 0) __builtin_amdgcn_s_wait_tensorcnt(0);
#endif
        __syncthreads();

        // A fragment: lane l16 = row, half selects K=0..15 / 16..31 (matches B)
        v16bf a;
        {
            const __bf16* ap = &sA[(wm * 16 + l16) * ASTR];
#pragma unroll
            for (int j = 0; j < 16; ++j) a[j] = ap[16 * half + j];
        }
        // B fragments: lane l16 = column, half selects K=0..15 / 16..31
        v16bf b0f, b1f;
        {
            const __bf16* bp0 = &sB[(wn * 32 + l16) * BSTR + 16 * half];
            const __bf16* bp1 = &sB[(wn * 32 + 16 + l16) * BSTR + 16 * half];
#pragma unroll
            for (int j = 0; j < 16; ++j) b0f[j] = bp0[j];
#pragma unroll
            for (int j = 0; j < 16; ++j) b1f[j] = bp1[j];
        }

        acc0 = __builtin_amdgcn_wmma_f32_16x16x32_bf16(false, a, false, b0f,
                                                       (short)0, acc0, false, false);
        acc1 = __builtin_amdgcn_wmma_f32_16x16x32_bf16(false, a, false, b1f,
                                                       (short)0, acc1, false, false);
        __syncthreads();
    }

    // epilogue: C layout — lane l16 = N, VGPR r -> M = r + 8*half
#pragma unroll
    for (int r = 0; r < 8; ++r) {
        int grow = m0 + wm * 16 + 8 * half + r;
        if (grow >= M) continue;
        int gc0 = n0 + wn * 32 + l16;
        if (gc0 < Nt) {
            float v = acc0[r] + bias[gc0];
            if (do_relu) v = fmaxf(v, 0.0f);
            C[(size_t)grow * ldc + gc0] = v;
        }
        int gc1 = gc0 + 16;
        if (gc1 < Nt) {
            float v = acc1[r] + bias[gc1];
            if (do_relu) v = fmaxf(v, 0.0f);
            C[(size_t)grow * ldc + gc1] = v;
        }
    }
}

// ---------------- launcher ----------------

extern "C" void kernel_launch(void* const* d_in, const int* in_sizes, int n_in,
                              void* d_out, int out_size, void* d_ws, size_t ws_size,
                              hipStream_t stream) {
    const float* x   = (const float*)d_in[0];
    const int*   ei  = (const int*)d_in[1];
    const float* Wf  = (const float*)d_in[2];
    const float* bft = (const float*)d_in[3];
    const float* W0  = (const float*)d_in[4];
    const float* b0  = (const float*)d_in[5];
    const float* g0  = (const float*)d_in[6];
    const float* be0 = (const float*)d_in[7];
    const float* W1  = (const float*)d_in[8];
    const float* b1  = (const float*)d_in[9];
    const float* g1  = (const float*)d_in[10];
    const float* be1 = (const float*)d_in[11];
    const float* Wc  = (const float*)d_in[12];
    const float* bc  = (const float*)d_in[13];
    const int* row = ei;
    const int* col = ei + NEDGES;

    float* ws = (float*)d_ws;
    size_t off = 0;
    auto alloc = [&](size_t nfloats) -> float* {
        float* p = ws + off;
        off += (nfloats + 63) & ~((size_t)63);
        return p;
    };
    float* deg    = alloc(NNODES);
    float* dis    = alloc(NNODES);
    float* nrm    = alloc(NEDGES);
    float* comb   = alloc((size_t)NNODES * 768);     // fp32 [h | agg1 | agg2]
    float* y      = alloc((size_t)NNODES * HID);     // pre-BN GEMM output
    float* bsum   = alloc(HID);
    float* bss    = alloc(HID);
    float* bscale = alloc(HID);
    float* bshift = alloc(HID);
    unsigned short* wfb    = (unsigned short*)alloc((size_t)IN_CH * HID / 2);
    unsigned short* w0b    = (unsigned short*)alloc((size_t)3 * HID * HID / 2);
    unsigned short* w1b    = (unsigned short*)alloc((size_t)3 * HID * HID / 2);
    unsigned short* wcb    = (unsigned short*)alloc((size_t)HID * BN / 2);      // Wc padded to 256x64
    unsigned short* xb     = (unsigned short*)alloc((size_t)NPAD * IN_CH / 2);  // bf16 x, row-padded
    unsigned short* combbf = (unsigned short*)alloc((size_t)NPAD * 768 / 2);    // bf16 combined, row-padded

    dim3 blk(256);

    // degrees -> norm
    k_zero<<<dim3((NNODES + 255) / 256), blk, 0, stream>>>(deg, NNODES);
    k_degree<<<dim3((NEDGES + 255) / 256), blk, 0, stream>>>(row, deg, NEDGES);
    k_dis<<<dim3((NNODES + 255) / 256), blk, 0, stream>>>(deg, dis, NNODES);
    k_norm<<<dim3((NEDGES + 255) / 256), blk, 0, stream>>>(row, col, dis, nrm, NEDGES);

    // operands -> bf16
    k_f2bf_arr<<<dim3((IN_CH * HID + 255) / 256), blk, 0, stream>>>(Wf, wfb, IN_CH * HID);
    k_f2bf_arr<<<dim3((3 * HID * HID + 255) / 256), blk, 0, stream>>>(W0, w0b, 3 * HID * HID);
    k_f2bf_arr<<<dim3((3 * HID * HID + 255) / 256), blk, 0, stream>>>(W1, w1b, 3 * HID * HID);
    // Wc: zero-pad 256x40 -> 256x64 so the GEMM B panel needs no bounds guard
    k_zero<<<dim3((HID * BN / 2 + 255) / 256), blk, 0, stream>>>((float*)wcb, HID * BN / 2);
    k_f2bf_pad<<<dim3((HID * OUT_CH + 255) / 256), blk, 0, stream>>>(Wc, wcb, OUT_CH, BN, HID * OUT_CH);
    k_f2bf_arr<<<dim3((NNODES * IN_CH + 255) / 256), blk, 0, stream>>>(x, xb, NNODES * IN_CH);

    // h0 = relu(x @ Wf + bft) -> combined[:,0:256]
    {
        dim3 g((NNODES + BM - 1) / BM, (HID + BN - 1) / BN);
        k_gemm_bf16<<<g, blk, 0, stream>>>(xb, IN_CH, wfb, HID, bft, comb, 768,
                                           NNODES, HID, IN_CH, 1);
    }

    for (int l = 0; l < 2; ++l) {
        const unsigned short* wb = l ? w1b : w0b;
        const float* bb = l ? b1 : b0;
        const float* gg = l ? g1 : g0;
        const float* be = l ? be1 : be0;

        long long aggN = (long long)NNODES * 512;
        k_zero_agg<<<dim3((unsigned)((aggN + 255) / 256)), blk, 0, stream>>>(comb, aggN);
        k_spmm<<<dim3((NEDGES + 7) / 8), blk, 0, stream>>>(comb, comb + 256, row, col, nrm, NEDGES);
        k_spmm<<<dim3((NEDGES + 7) / 8), blk, 0, stream>>>(comb + 256, comb + 512, row, col, nrm, NEDGES);

        // combined -> bf16 mirror, then y = combined @ W + b
        long long cn = (long long)NNODES * 768;
        k_comb2bf<<<dim3((unsigned)((cn + 255) / 256)), blk, 0, stream>>>(comb, combbf, cn);
        dim3 g((NNODES + BM - 1) / BM, (HID + BN - 1) / BN);
        k_gemm_bf16<<<g, blk, 0, stream>>>(combbf, 768, wb, HID, bb, y, HID,
                                           NNODES, HID, 3 * HID, 0);

        // batchnorm (training stats) + relu -> combined[:,0:256]
        k_zero<<<dim3(2), blk, 0, stream>>>(bsum, 512);
        k_bn_reduce<<<dim3((NNODES + 127) / 128), blk, 0, stream>>>(y, bsum, bss, NNODES);
        k_bn_final<<<dim3(1), blk, 0, stream>>>(bsum, bss, gg, be, bscale, bshift, NNODES);
        long long hn = (long long)NNODES * HID;
        k_bn_apply<<<dim3((unsigned)((hn + 255) / 256)), blk, 0, stream>>>(y, bscale, bshift, comb, hn);
    }

    // final h -> bf16 (slice 0 of combbf, stride 768), out = h @ Wc + bc
    {
        long long cn = (long long)NNODES * 768;
        k_comb2bf<<<dim3((unsigned)((cn + 255) / 256)), blk, 0, stream>>>(comb, combbf, cn);
        dim3 g((NNODES + BM - 1) / BM, (OUT_CH + BN - 1) / BN);
        k_gemm_bf16<<<g, blk, 0, stream>>>(combbf, 768, wcb, BN, bc, (float*)d_out, OUT_CH,
                                           NNODES, OUT_CH, HID, 0);
    }
}
